// ComponentGNN_73778948211181
// MI455X (gfx1250) — compile-verified, matrix operand order checked by reading, not run on hardware
//
#include <hip/hip_runtime.h>
#include <hip/hip_bf16.h>

#define HEADS 4
#define HID 64
#define NGRAPH 64
#define NEG_SLOPE_C 0.2f
#define EPS_C 1e-5f

typedef __attribute__((ext_vector_type(16))) __bf16 v16bf;
typedef __attribute__((ext_vector_type(8)))  float  v8f;

struct __attribute__((aligned(16))) U4 { unsigned int x, y, z, w; };
union BfFrag { v16bf v; U4 q[2]; };

__device__ inline unsigned short f2bf(float f) {
    unsigned int u = __float_as_uint(f);
    u = u + 0x7fffu + ((u >> 16) & 1u);          // round-to-nearest-even
    return (unsigned short)(u >> 16);
}
__device__ inline int f2ord(float f) {           // order-preserving float->int
    int i = __float_as_int(f);
    return i >= 0 ? i : (i ^ 0x7fffffff);
}
__device__ inline float ord2f(int i) {
    return __int_as_float(i >= 0 ? i : (i ^ 0x7fffffff));
}

// ---------------------------------------------------------------------------
// Tiled bf16-WMMA GEMM: out[M,F] = act(A[M,K] @ W[K,F] + bias)
// 64-row blocks, 128 threads = 4 waves; wave w computes rows [w*16, w*16+16).
// W staged transposed (F-major) in LDS once; A staged in KC-wide chunks.
// ---------------------------------------------------------------------------
template<int K, int F, int KC, bool RELU>
__global__ __launch_bounds__(128) void gemm_wmma(const float* __restrict__ A,
                                                 const float* __restrict__ W,
                                                 const float* __restrict__ bias,
                                                 float* __restrict__ out, int M)
{
    __shared__ unsigned short Wt[F * K];      // W transposed, bf16
    __shared__ unsigned short As[64 * KC];    // A chunk, bf16

    const int tid   = threadIdx.x;
    const int lane  = tid & 31;
    const int wave  = tid >> 5;
    const int r     = lane & 15;              // row/col within 16-tile
    const int hhalf = lane >> 4;              // lane-half selects K sub-block
    const int mBase = blockIdx.x * 64;

    // Stage W transposed + converted (L2-hot, tiny).
    for (int idx = tid; idx < F * K; idx += 128) {
        int f = idx / K, k = idx - f * K;
        Wt[idx] = f2bf(W[(size_t)k * F + f]);
    }

    v8f acc[F / 16];
    #pragma unroll
    for (int i = 0; i < F / 16; ++i) acc[i] = v8f{};

    for (int kc = 0; kc < K; kc += KC) {
        __syncthreads();
        for (int idx = tid; idx < 64 * KC; idx += 128) {
            int rr = idx / KC, kk = idx - rr * KC;
            int row = mBase + rr;
            As[idx] = (row < M) ? f2bf(A[(size_t)row * K + kc + kk]) : (unsigned short)0;
        }
        __syncthreads();
        #pragma unroll
        for (int k0 = 0; k0 < KC; k0 += 32) {
            // A fragment: lanes 0-15 row M=r with K {k0..k0+7, k0+16..k0+23},
            // lanes 16-31 same rows, K sets shifted by 8.
            BfFrag a;
            const unsigned short* ap = &As[(wave * 16 + r) * KC + k0 + hhalf * 8];
            a.q[0] = *(const U4*)(ap);
            a.q[1] = *(const U4*)(ap + 16);
            #pragma unroll
            for (int nt = 0; nt < F / 16; ++nt) {
                BfFrag b;   // B fragment mirrors A with N in place of M (Wt is F-major)
                const unsigned short* bp = &Wt[(nt * 16 + r) * K + (kc + k0) + hhalf * 8];
                b.q[0] = *(const U4*)(bp);
                b.q[1] = *(const U4*)(bp + 16);
                acc[nt] = __builtin_amdgcn_wmma_f32_16x16x32_bf16(
                    false, a.v, false, b.v, (short)0, acc[nt], false, false);
            }
        }
    }

    // C/D layout: VGPR j -> M = j (lanes 0-15) / j+8 (lanes 16-31), N = r.
    const int mRow0 = mBase + wave * 16 + hhalf * 8;
    #pragma unroll
    for (int nt = 0; nt < F / 16; ++nt) {
        int col = nt * 16 + r;
        float bcol = bias[col];
        #pragma unroll
        for (int j = 0; j < 8; ++j) {
            int row = mRow0 + j;
            if (row < M) {
                float val = acc[nt][j] + bcol;
                if (RELU) val = fmaxf(val, 0.f);
                out[(size_t)row * F + col] = val;
            }
        }
    }
}

// ---------------------------------------------------------------------------
// Edge pass 1: logits[e,h] = dot(leakyrelu(xl[src]+xr[dst]), att[h]); seg-max.
// ---------------------------------------------------------------------------
template<int C>
__global__ void edge_logits_kernel(const float* __restrict__ xl, const float* __restrict__ xr,
                                   const int* __restrict__ src, const int* __restrict__ dst,
                                   const float* __restrict__ att,
                                   float* __restrict__ logits, int* __restrict__ nmax, int E)
{
    int t = blockIdx.x * blockDim.x + threadIdx.x;
    if (t >= E * HEADS) return;
    int e = t >> 2, h = t & 3;
    int s = src[e], d = dst[e];
    const float* pl = xl + ((size_t)s * HEADS + h) * C;
    const float* pr = xr + ((size_t)d * HEADS + h) * C;
    const float* pa = att + h * C;
    float accv = 0.f;
    #pragma unroll
    for (int c = 0; c < C; ++c) {
        float z = pl[c] + pr[c];
        z = z > 0.f ? z : NEG_SLOPE_C * z;
        accv += z * pa[c];
    }
    logits[(size_t)e * HEADS + h] = accv;
    atomicMax(&nmax[(size_t)d * HEADS + h], f2ord(accv));
}

// ---------------------------------------------------------------------------
// Edge pass 2 (fused): ev = exp(logit - max[dst]); denom += ev; acc += ev*xl[src]
// ---------------------------------------------------------------------------
template<int C>
__global__ void edge_accum_kernel(const float* __restrict__ xl,
                                  const int* __restrict__ src, const int* __restrict__ dst,
                                  const float* __restrict__ logits, const int* __restrict__ nmax,
                                  float* __restrict__ denom, float* __restrict__ accum, int E)
{
    int t = blockIdx.x * blockDim.x + threadIdx.x;
    if (t >= E * HEADS) return;
    int e = t >> 2, h = t & 3;
    int s = src[e], d = dst[e];
    float m  = ord2f(nmax[(size_t)d * HEADS + h]);
    float ev = __expf(logits[(size_t)e * HEADS + h] - m);
    atomicAdd(&denom[(size_t)d * HEADS + h], ev);
    const float* pl = xl + ((size_t)s * HEADS + h) * C;
    float* pd = accum + ((size_t)d * HEADS + h) * C;
    for (int c = 0; c < C; ++c)
        atomicAdd(&pd[c], ev * pl[c]);
}

// Node pass, layer 0 (concat heads): normalize, +bias, BN(eval), ReLU.
__global__ void finalize_concat_kernel(const float* __restrict__ accum, const float* __restrict__ denom,
                                       const float* __restrict__ bias,
                                       const float* __restrict__ gamma, const float* __restrict__ beta,
                                       const float* __restrict__ mean, const float* __restrict__ var,
                                       float* __restrict__ out, int Nn)
{
    long long t = (long long)blockIdx.x * blockDim.x + threadIdx.x;
    if (t >= (long long)Nn * HID) return;
    int n = (int)(t / HID), col = (int)(t - (long long)n * HID);
    int h = col >> 4;   // 16 channels per head
    float d = denom[(size_t)n * HEADS + h];
    float val = d > 0.f ? accum[t] / d : 0.f;
    val += bias[col];
    val = (val - mean[col]) * rsqrtf(var[col] + EPS_C) * gamma[col] + beta[col];
    out[t] = fmaxf(val, 0.f);
}

// Node pass, layer 1 (mean over heads): normalize, mean, +bias, BN(eval), ReLU.
__global__ void finalize_mean_kernel(const float* __restrict__ accum, const float* __restrict__ denom,
                                     const float* __restrict__ bias,
                                     const float* __restrict__ gamma, const float* __restrict__ beta,
                                     const float* __restrict__ mean, const float* __restrict__ var,
                                     float* __restrict__ out, int Nn)
{
    long long t = (long long)blockIdx.x * blockDim.x + threadIdx.x;
    if (t >= (long long)Nn * HID) return;
    int n = (int)(t / HID), col = (int)(t - (long long)n * HID);
    float s = 0.f;
    #pragma unroll
    for (int h = 0; h < HEADS; ++h) {
        float d = denom[(size_t)n * HEADS + h];
        float a = accum[((size_t)n * HEADS + h) * HID + col];
        s += d > 0.f ? a / d : 0.f;
    }
    float val = s * 0.25f + bias[col];
    val = (val - mean[col]) * rsqrtf(var[col] + EPS_C) * gamma[col] + beta[col];
    out[t] = fmaxf(val, 0.f);
}

__global__ void fill_f32_kernel(float* p, float v, long long n) {
    long long t = (long long)blockIdx.x * blockDim.x + threadIdx.x;
    if (t < n) p[t] = v;
}
__global__ void fill_i32_kernel(int* p, int v, long long n) {
    long long t = (long long)blockIdx.x * blockDim.x + threadIdx.x;
    if (t < n) p[t] = v;
}

__global__ void pool_kernel(const float* __restrict__ hin, const int* __restrict__ batch,
                            float* __restrict__ sums, float* __restrict__ cnts, int Nn)
{
    long long t = (long long)blockIdx.x * blockDim.x + threadIdx.x;
    if (t >= (long long)Nn * HID) return;
    int n = (int)(t / HID), c = (int)(t - (long long)n * HID);
    int g = batch[n];
    atomicAdd(&sums[(size_t)g * HID + c], hin[t]);
    if (c == 0) atomicAdd(&cnts[g], 1.0f);
}

__global__ void classifier_kernel(const float* __restrict__ sums, const float* __restrict__ cnts,
                                  const float* __restrict__ Wc1, const float* __restrict__ bc1,
                                  const float* __restrict__ Wc2, const float* __restrict__ bc2,
                                  float* __restrict__ out)
{
    int g = threadIdx.x;
    if (g >= NGRAPH) return;
    float inv = 1.f / fmaxf(cnts[g], 1.f);
    float hv[32];
    #pragma unroll
    for (int j = 0; j < 32; ++j) {
        float a = bc1[j];
        for (int c = 0; c < HID; ++c) a += sums[g * HID + c] * inv * Wc1[c * 32 + j];
        hv[j] = fmaxf(a, 0.f);
    }
    #pragma unroll
    for (int k = 0; k < 2; ++k) {
        float o = bc2[k];
        for (int j = 0; j < 32; ++j) o += hv[j] * Wc2[j * 2 + k];
        out[g * 2 + k] = o;
    }
}

extern "C" void kernel_launch(void* const* d_in, const int* in_sizes, int n_in,
                              void* d_out, int out_size, void* d_ws, size_t ws_size,
                              hipStream_t stream)
{
    const float* x     = (const float*)d_in[0];
    const int*   edge  = (const int*)d_in[1];
    const int*   batch = (const int*)d_in[2];
    const float* W_in  = (const float*)d_in[3];
    const float* b_in  = (const float*)d_in[4];
    const float* Wl0   = (const float*)d_in[5];
    const float* bl0   = (const float*)d_in[6];
    const float* Wr0   = (const float*)d_in[7];
    const float* br0   = (const float*)d_in[8];
    const float* att0  = (const float*)d_in[9];
    const float* bias0 = (const float*)d_in[10];
    const float* g0    = (const float*)d_in[11];
    const float* be0   = (const float*)d_in[12];
    const float* m0    = (const float*)d_in[13];
    const float* v0    = (const float*)d_in[14];
    const float* Wl1   = (const float*)d_in[15];
    const float* bl1   = (const float*)d_in[16];
    const float* Wr1   = (const float*)d_in[17];
    const float* br1   = (const float*)d_in[18];
    const float* att1  = (const float*)d_in[19];
    const float* bias1 = (const float*)d_in[20];
    const float* g1    = (const float*)d_in[21];
    const float* be1   = (const float*)d_in[22];
    const float* m1    = (const float*)d_in[23];
    const float* v1    = (const float*)d_in[24];
    const float* Wc1   = (const float*)d_in[25];
    const float* bc1   = (const float*)d_in[26];
    const float* Wc2   = (const float*)d_in[27];
    const float* bc2   = (const float*)d_in[28];

    const int M = in_sizes[2];        // nodes
    const int E = in_sizes[1] / 2;    // edges
    const int* srcIdx = edge;
    const int* dstIdx = edge + E;

    // Workspace layout
    float* ws      = (float*)d_ws;
    float* hA      = ws;                            // M*64
    float* xl      = hA      + (size_t)M * 64;      // M*256
    float* xr      = xl      + (size_t)M * 256;     // M*256
    float* accB    = xr      + (size_t)M * 256;     // M*256
    float* hB      = accB    + (size_t)M * 256;     // M*64
    float* hC      = hB      + (size_t)M * 64;      // M*64
    float* logitsE = hC      + (size_t)M * 64;      // E*4
    float* denom   = logitsE + (size_t)E * 4;       // M*4
    int*   nmax    = (int*)  (denom + (size_t)M * 4);   // M*4
    float* psum    = (float*)(nmax  + (size_t)M * 4);   // 64*64
    float* pcnt    = psum + NGRAPH * HID;               // 64

    const unsigned mB = (unsigned)((M + 63) / 64);
    const long long nM4 = (long long)M * 4, nM64 = (long long)M * 64, nM256 = (long long)M * 256;
    const long long nE4 = (long long)E * 4;
    auto blk = [](long long n) { return (unsigned)((n + 255) / 256); };

    // h = relu(x @ W_in + b_in)
    gemm_wmma<256, 64, 128, true><<<mB, 128, 0, stream>>>(x, W_in, b_in, hA, M);

    // ---- GATv2 layer 0 (heads=4, ch=16, concat) ----
    gemm_wmma<64, 64, 64, false><<<mB, 128, 0, stream>>>(hA, Wl0, bl0, xl, M);
    gemm_wmma<64, 64, 64, false><<<mB, 128, 0, stream>>>(hA, Wr0, br0, xr, M);
    fill_i32_kernel<<<blk(nM4), 256, 0, stream>>>(nmax, (int)0x80000000, nM4);
    fill_f32_kernel<<<blk(nM4), 256, 0, stream>>>(denom, 0.f, nM4);
    fill_f32_kernel<<<blk(nM64), 256, 0, stream>>>(accB, 0.f, nM64);
    edge_logits_kernel<16><<<blk(nE4), 256, 0, stream>>>(xl, xr, srcIdx, dstIdx, att0, logitsE, nmax, E);
    edge_accum_kernel<16><<<blk(nE4), 256, 0, stream>>>(xl, srcIdx, dstIdx, logitsE, nmax, denom, accB, E);
    finalize_concat_kernel<<<blk(nM64), 256, 0, stream>>>(accB, denom, bias0, g0, be0, m0, v0, hB, M);

    // ---- GATv2 layer 1 (heads=4, ch=64, mean) ----
    gemm_wmma<64, 256, 64, false><<<mB, 128, 0, stream>>>(hB, Wl1, bl1, xl, M);
    gemm_wmma<64, 256, 64, false><<<mB, 128, 0, stream>>>(hB, Wr1, br1, xr, M);
    fill_i32_kernel<<<blk(nM4), 256, 0, stream>>>(nmax, (int)0x80000000, nM4);
    fill_f32_kernel<<<blk(nM4), 256, 0, stream>>>(denom, 0.f, nM4);
    fill_f32_kernel<<<blk(nM256), 256, 0, stream>>>(accB, 0.f, nM256);
    edge_logits_kernel<64><<<blk(nE4), 256, 0, stream>>>(xl, xr, srcIdx, dstIdx, att1, logitsE, nmax, E);
    edge_accum_kernel<64><<<blk(nE4), 256, 0, stream>>>(xl, srcIdx, dstIdx, logitsE, nmax, denom, accB, E);
    finalize_mean_kernel<<<blk(nM64), 256, 0, stream>>>(accB, denom, bias1, g1, be1, m1, v1, hC, M);

    // ---- global mean pool + classifier ----
    fill_f32_kernel<<<blk(NGRAPH * HID + NGRAPH), 256, 0, stream>>>(psum, 0.f, NGRAPH * HID + NGRAPH);
    pool_kernel<<<blk(nM64), 256, 0, stream>>>(hC, batch, psum, pcnt, M);
    classifier_kernel<<<1, 64, 0, stream>>>(psum, pcnt, Wc1, bc1, Wc2, bc2, (float*)d_out);
}
